// CGGNN_40724879900769
// MI455X (gfx1250) — matrix operand order repaired and checked by reading, not compiled, not run on hardware
//
#include <hip/hip_runtime.h>
#include <hip/hip_bf16.h>
#include <stdint.h>
#include <stddef.h>

// ---------------- problem constants (fixed by setup_inputs) ----------------
#define NN     10000      // nodes
#define EE     160000     // edges
#define GG     500        // graphs
#define HID    64         // hidden channels
#define NRBF   20
#define NLAYER 4
#define EPSV   1e-6f
#define RSQRT2 0.70710678118654752440f

// padded K strides (multiples of 32 -> guard-free WMMA K loop)
#define KP_EDGE 96        // logical 84
#define KP_X0   96        // logical 65
#define KP_NODE 128       // logical 128

typedef __attribute__((ext_vector_type(16))) __bf16 v16bf;
typedef __attribute__((ext_vector_type(8)))  float  v8f;

// ---------------- Clifford algebra tables (Cl(3,0,0)) ----------------
// blade order: 1, e1, e2, e3, e12, e13, e23, e123  (masks below)
constexpr int BLADEMASK[8] = {0, 1, 2, 4, 3, 5, 6, 7};
constexpr int IDXOF[8]     = {0, 1, 2, 4, 3, 5, 6, 7};  // mask -> blade index
constexpr int GR8[8]       = {0, 1, 1, 1, 2, 2, 2, 3};  // grade per blade

__host__ __device__ constexpr int cl_sign(int a, int b) {
  int t = 0, s = a >> 1;
  while (s) { t += __builtin_popcount(s & b); s >>= 1; }
  return (t & 1) ? -1 : 1;
}

// PATH_LUT[gi][gj][gk] -> gp_w path index (argwhere order), -1 if no path
constexpr int PLUT[4][4][4] = {
  { { 0,-1,-1,-1}, {-1, 1,-1,-1}, {-1,-1, 2,-1}, {-1,-1,-1, 3} },
  { {-1, 4,-1,-1}, { 5,-1, 6,-1}, {-1, 7,-1, 8}, {-1,-1, 9,-1} },
  { {-1,-1,10,-1}, {-1,11,-1,12}, {13,-1,14,-1}, {-1,15,-1,-1} },
  { {-1,-1,-1,16}, {-1,-1,17,-1}, {-1,18,-1,-1}, {19,-1,-1,-1} },
};

// ---------------- small helpers ----------------
// pack two f32 into packed bf16 dword: round-half-up + single v_perm_b32
__device__ __forceinline__ unsigned int pack2bf(float x, float y) {
  unsigned int ux = __float_as_uint(x) + 0x8000u;
  unsigned int uy = __float_as_uint(y) + 0x8000u;
  // result bytes: [0,1]=ux bytes[2,3], [2,3]=uy bytes[2,3]  ({a=uy, b=ux} byte pool)
  return __builtin_amdgcn_perm(uy, ux, 0x07060302u);
}
__device__ __forceinline__ unsigned short f2bf1(float f) {
  return (unsigned short)((__float_as_uint(f) + 0x8000u) >> 16);
}
__device__ __forceinline__ float sigm(float x) { return 1.0f / (1.0f + __expf(-x)); }

static inline int cdiv(int a, int b) { return (a + b - 1) / b; }

// ---------------- generic zero ----------------
__global__ void k_zero(float* __restrict__ p, int n) {
  int i = blockIdx.x * 256 + threadIdx.x;
  if (i < n) p[i] = 0.0f;
}

// --------- weight prep: f32 [Cout][K][4] (or [Cout][K]) -> bf16 [4][KP][64], zero-padded
__global__ void k_prepw(const float* __restrict__ w, unsigned short* __restrict__ wb,
                        int K, int KP, int subspaces) {
  int idx = blockIdx.x * 256 + threadIdx.x;
  int tot = 4 * KP * 64;
  if (idx >= tot) return;
  int g = idx / (KP * 64);
  int r = idx % (KP * 64);
  int k = r / 64, n = r % 64;
  float v = 0.0f;
  if (k < K) v = subspaces ? w[((size_t)n * K + k) * 4 + g] : w[(size_t)n * K + k];
  wb[idx] = f2bf1(v);
}

// ---------------- WMMA GEMM: Out[bl][m][n] = sum_k X[bl][m][k] * W[grade(bl)][k][n]
// X f32 blade-major [8][M][KP] (pad columns are zero); W bf16 [4][KP][64];
// Out f32 [8][M][64]; bias (nullptr ok) added to blade 0 only.
// Block tile: 256 rows x 64 cols; each of 8 waves owns 32 rows (2 A-fragments),
// so every B fragment read from LDS feeds two WMMAs.
__global__ __launch_bounds__(256)
void k_gemm(const float* __restrict__ X, const unsigned short* __restrict__ W,
            const float* __restrict__ bias, float* __restrict__ Out, int M, int KP) {
  constexpr int GRD[8] = {0, 1, 1, 1, 2, 2, 2, 3};
  const int blade = blockIdx.z;
  const int g = GRD[blade];
  const float* Xb = X + (size_t)blade * M * KP;
  float* Ob = Out + (size_t)blade * M * 64;
  const unsigned short* Wg = W + (size_t)g * KP * 64;

  // A tile: 256 rows x 32 k, halfword stride 40 (8B-aligned rows, conflict-free frags)
  __shared__ unsigned short sA[256 * 40];
  // B tile transposed [n][k], stride 36
  __shared__ unsigned short sB[64 * 36];

  const int tid  = threadIdx.x;
  const int wave = tid >> 5;
  const int lane = tid & 31;
  const int m0   = blockIdx.x * 256;
  const int mw   = m0 + wave * 32;

  v8f acc[2][4];
  #pragma unroll
  for (int s = 0; s < 2; ++s)
    #pragma unroll
    for (int t = 0; t < 4; ++t)
      acc[s][t] = (v8f){0.f,0.f,0.f,0.f,0.f,0.f,0.f,0.f};

  union Frag { v16bf v; unsigned int u[8]; };

  for (int k0 = 0; k0 < KP; k0 += 32) {
    // ---- A tile: staged global_load_b128 (4 in flight), then pack+store ----
    #pragma unroll
    for (int half = 0; half < 2; ++half) {
      float4 va[4];
      #pragma unroll
      for (int it = 0; it < 4; ++it) {
        int idx = tid + (half * 4 + it) * 256;   // 2048 float4 slots = 256 rows x 8
        int r   = idx >> 3;
        int c4  = (idx & 7) * 4;
        int gm  = m0 + r;
        int gmc = gm < M ? gm : M - 1;
        va[it] = *(const float4*)(Xb + (size_t)gmc * KP + k0 + c4);
      }
      #pragma unroll
      for (int it = 0; it < 4; ++it) {
        int idx = tid + (half * 4 + it) * 256;
        int r   = idx >> 3;
        int c4  = (idx & 7) * 4;
        float4 v = va[it];
        if (m0 + r >= M) { v.x = 0.f; v.y = 0.f; v.z = 0.f; v.w = 0.f; }
        uint2 pk;
        pk.x = pack2bf(v.x, v.y);
        pk.y = pack2bf(v.z, v.w);
        *(uint2*)(sA + r * 40 + c4) = pk;        // one 8B LDS store
      }
    }
    // ---- B tile: staged dword loads from bf16 panel, transpose into [n][k] ----
    {
      unsigned int wd[4];
      #pragma unroll
      for (int it = 0; it < 4; ++it) {
        int idx = tid + it * 256;                // 1024 dwords = 32 k x 32 n-pairs
        int kk = idx >> 5, nh = idx & 31;
        wd[it] = *(const unsigned int*)(Wg + (size_t)(k0 + kk) * 64 + nh * 2);
      }
      #pragma unroll
      for (int it = 0; it < 4; ++it) {
        int idx = tid + it * 256;
        int kk = idx >> 5, nh = idx & 31;
        sB[(nh * 2    ) * 36 + kk] = (unsigned short)(wd[it] & 0xFFFFu);
        sB[(nh * 2 + 1) * 36 + kk] = (unsigned short)(wd[it] >> 16);
      }
    }
    if (k0 + 32 < KP)  // CDNA5 global_prefetch_b8 for next K panel
      __builtin_prefetch(Xb + (size_t)(m0 + (tid & 255)) * KP + k0 + 32, 0, 1);
    __syncthreads();

    // A fragments (16-bit A 16x32 layout, ISA 7.12.2): rows mw..mw+15 and mw+16..mw+31
    Frag a0, a1;
    const int am = wave * 32 + (lane & 15);
    const int kh = (lane >> 4) * 8;
    #pragma unroll
    for (int v = 0; v < 4; ++v) {
      a0.u[v]     = *(const unsigned int*)(sA + am * 40 + kh + 2 * v);
      a0.u[4 + v] = *(const unsigned int*)(sA + am * 40 + 16 + kh + 2 * v);
      a1.u[v]     = *(const unsigned int*)(sA + (am + 16) * 40 + kh + 2 * v);
      a1.u[4 + v] = *(const unsigned int*)(sA + (am + 16) * 40 + 16 + kh + 2 * v);
    }

    // B fragments + WMMA over the 4 column tiles (N=64); each B feeds two WMMAs
    const int khB = (lane >> 4) * 8;
    #pragma unroll
    for (int nt = 0; nt < 4; ++nt) {
      Frag b;
      const int n = nt * 16 + (lane & 15);
      #pragma unroll
      for (int v = 0; v < 8; ++v)
        b.u[v] = *(const unsigned int*)(sB + n * 36 + khB + ((v >> 2) << 4) + 2 * (v & 3));
      acc[0][nt] = __builtin_amdgcn_wmma_f32_16x16x32_bf16(
          false, a0.v, false, b.v, (short)0, acc[0][nt], false, false);
      acc[1][nt] = __builtin_amdgcn_wmma_f32_16x16x32_bf16(
          false, a1.v, false, b.v, (short)0, acc[1][nt], false, false);
    }
    __syncthreads();
  }

  // epilogue: 16x16 f32 C/D layout (VGPR j -> rows j / j+8), two row strips
  #pragma unroll
  for (int s = 0; s < 2; ++s) {
    #pragma unroll
    for (int nt = 0; nt < 4; ++nt) {
      const int n = nt * 16 + (lane & 15);
      #pragma unroll
      for (int j = 0; j < 8; ++j) {
        int r = mw + s * 16 + j + ((lane >> 4) << 3);
        if (r < M) {
          float v = acc[s][nt][j];
          if (bias != nullptr && blade == 0) v += bias[n];
          Ob[(size_t)r * 64 + n] = v;
        }
      }
    }
  }
}

// ---------------- mv_silu (in place), x: [8][B][64] ----------------
__global__ void k_silu(float* __restrict__ x, const float* __restrict__ a,
                       const float* __restrict__ b, int B) {
  int idx = blockIdx.x * 256 + threadIdx.x;
  if (idx >= B * 64) return;
  int c = idx & 63;
  size_t bs = (size_t)B * 64;
  float v[8];
  #pragma unroll
  for (int i = 0; i < 8; ++i) v[i] = x[(size_t)i * bs + idx];
  float inv[4];
  inv[0] = v[0];
  inv[1] = v[1]*v[1] + v[2]*v[2] + v[3]*v[3];
  inv[2] = v[4]*v[4] + v[5]*v[5] + v[6]*v[6];
  inv[3] = v[7]*v[7];
  float f[4];
  #pragma unroll
  for (int g = 0; g < 4; ++g) f[g] = sigm(a[c*4+g] * inv[g] + b[c*4+g]);
  #pragma unroll
  for (int i = 0; i < 8; ++i) x[(size_t)i * bs + idx] = v[i] * f[GR8[i]];
}

// ---------------- mv_normalize (in place) ----------------
__global__ void k_mvnorm(float* __restrict__ x, const float* __restrict__ na, int B) {
  int idx = blockIdx.x * 256 + threadIdx.x;
  if (idx >= B * 64) return;
  int c = idx & 63;
  size_t bs = (size_t)B * 64;
  float v[8];
  #pragma unroll
  for (int i = 0; i < 8; ++i) v[i] = x[(size_t)i * bs + idx];
  float q[4];
  q[0] = v[0]*v[0];
  q[1] = v[1]*v[1] + v[2]*v[2] + v[3]*v[3];
  q[2] = v[4]*v[4] + v[5]*v[5] + v[6]*v[6];
  q[3] = v[7]*v[7];
  float den[4];
  #pragma unroll
  for (int g = 0; g < 4; ++g) {
    float s = sigm(na[c*4+g]);
    den[g] = s * (sqrtf(q[g]) - 1.0f) + 1.0f + EPSV;
  }
  #pragma unroll
  for (int i = 0; i < 8; ++i) x[(size_t)i * bs + idx] = v[i] / den[GR8[i]];
}

// ---------------- geometric product + combine: out = (left + geo)/sqrt(2) ---
// t,xr: [8][B][64]; out holds `left` on entry (from w_left GEMM)
__global__ void k_geo(const float* __restrict__ t, const float* __restrict__ xr,
                      float* __restrict__ out, const float* __restrict__ gpw, int B) {
  int idx = blockIdx.x * 256 + threadIdx.x;
  if (idx >= B * 64) return;
  int c = idx & 63;
  size_t bs = (size_t)B * 64;
  float x[8], y[8], geo[8];
  #pragma unroll
  for (int i = 0; i < 8; ++i) { x[i] = t[(size_t)i*bs + idx]; y[i] = xr[(size_t)i*bs + idx]; geo[i] = 0.0f; }
  float w20[20];
  #pragma unroll
  for (int p = 0; p < 20; ++p) w20[p] = gpw[c * 20 + p];
  #pragma unroll
  for (int i = 0; i < 8; ++i) {
    #pragma unroll
    for (int k = 0; k < 8; ++k) {
      const int j  = IDXOF[BLADEMASK[i] ^ BLADEMASK[k]];
      const int p  = PLUT[GR8[i]][GR8[j]][GR8[k]];
      const float s = (float)cl_sign(BLADEMASK[i], BLADEMASK[k]);
      geo[j] += s * w20[p] * x[i] * y[k];
    }
  }
  #pragma unroll
  for (int j = 0; j < 8; ++j) {
    size_t o = (size_t)j * bs + idx;
    out[o] = (out[o] + geo[j]) * RSQRT2;
  }
}

// ---------------- mv_layernorm (in place): one block per sample b ----------
__global__ void k_ln(float* __restrict__ x, const float* __restrict__ la, int B) {
  __shared__ float red[64];
  int b = blockIdx.x, c = threadIdx.x;
  size_t bs = (size_t)B * 64;
  size_t base = (size_t)b * 64 + c;
  float v[8], ss = 0.0f;
  #pragma unroll
  for (int i = 0; i < 8; ++i) { v[i] = x[(size_t)i * bs + base]; ss += v[i] * v[i]; }
  red[c] = sqrtf(ss);
  __syncthreads();
  for (int s = 32; s > 0; s >>= 1) { if (c < s) red[c] += red[c + s]; __syncthreads(); }
  float mean = red[0] * (1.0f / 64.0f) + EPSV;
  float sc = la[c] / mean;
  #pragma unroll
  for (int i = 0; i < 8; ++i) x[(size_t)i * bs + base] = v[i] * sc;
}

// ---------------- graph preprocessing ----------------
__global__ void k_count_edges(const int* __restrict__ rows, float* __restrict__ cnt) {
  int e = blockIdx.x * 256 + threadIdx.x;
  if (e < EE) atomicAdd(&cnt[rows[e]], 1.0f);
}
__global__ void k_nodestats(const float* __restrict__ pos, const int* __restrict__ batch,
                            float* __restrict__ cntG, float* __restrict__ psum) {
  int n = blockIdx.x * 256 + threadIdx.x;
  if (n >= NN) return;
  int g = batch[n];
  atomicAdd(&cntG[g], 1.0f);
  atomicAdd(&psum[g*3+0], pos[n*3+0]);
  atomicAdd(&psum[g*3+1], pos[n*3+1]);
  atomicAdd(&psum[g*3+2], pos[n*3+2]);
}
__global__ void k_pmean(const float* __restrict__ psum, const float* __restrict__ cntG,
                        float* __restrict__ pmean) {
  int i = blockIdx.x * 256 + threadIdx.x;
  if (i < GG * 3) pmean[i] = psum[i] / fmaxf(cntG[i / 3], 1.0f);
}
__global__ void k_rbf(const float* __restrict__ pos, const int* __restrict__ rows,
                      const int* __restrict__ cols, float* __restrict__ rbf) {
  int e = blockIdx.x * 256 + threadIdx.x;
  if (e >= EE) return;
  int r = rows[e], c = cols[e];
  float dx = pos[r*3+0]-pos[c*3+0], dy = pos[r*3+1]-pos[c*3+1], dz = pos[r*3+2]-pos[c*3+2];
  float d = sqrtf(dx*dx + dy*dy + dz*dz);
  const float step = 10.0f / 19.0f;   // linspace(0,10,20)
  #pragma unroll
  for (int j = 0; j < NRBF; ++j) {
    float t = d - step * j;
    rbf[(size_t)e * NRBF + j] = __expf(-2.0f * t * t);  // width=0.5 -> 1/(2w^2)=2
  }
}
// initial features x0: [8][NN][KP_X0] (channels: 0..63 scalar embed, 64 vector, pad 0)
__global__ void k_x0(const float* __restrict__ emb, const int* __restrict__ z,
                     const float* __restrict__ pos, const float* __restrict__ pmean,
                     const int* __restrict__ batch, float* __restrict__ x0) {
  int idx = blockIdx.x * 256 + threadIdx.x;
  if (idx >= NN * KP_X0) return;
  int n = idx / KP_X0, c = idx % KP_X0;
  size_t bs = (size_t)NN * KP_X0;
  #pragma unroll
  for (int i = 0; i < 8; ++i) {
    float v = 0.0f;
    if (c < 64) { if (i == 0) v = emb[(size_t)z[n] * 64 + c]; }
    else if (c == 64 && i >= 1 && i <= 3) v = pos[n*3 + (i-1)] - pmean[batch[n]*3 + (i-1)];
    x0[(size_t)i * bs + idx] = v;
  }
}

// ---------------- edge / node tensor assembly ----------------
__global__ void k_build_edge(const float* __restrict__ h, const float* __restrict__ rbf,
                             const int* __restrict__ rows, const int* __restrict__ cols,
                             float* __restrict__ Xe) {
  int idx = blockIdx.x * 256 + threadIdx.x;
  if (idx >= EE * KP_EDGE) return;
  int e = idx / KP_EDGE, c = idx % KP_EDGE;
  int r = rows[e], cl = cols[e];
  size_t hs = (size_t)NN * 64, es = (size_t)EE * KP_EDGE;
  #pragma unroll
  for (int i = 0; i < 8; ++i) {
    float v = 0.0f;
    if (c < 64)      v = h[(size_t)i*hs + (size_t)r*64 + c] - h[(size_t)i*hs + (size_t)cl*64 + c];
    else if (c < 84) v = (i == 0) ? rbf[(size_t)e * NRBF + (c - 64)] : 0.0f;
    Xe[(size_t)i * es + idx] = v;
  }
}
__global__ void k_scatter_msg(const float* __restrict__ msg, const int* __restrict__ rows,
                              float* __restrict__ agg) {
  int idx = blockIdx.x * 256 + threadIdx.x;
  if (idx >= EE * 64) return;
  int e = idx >> 6, c = idx & 63;
  int r = rows[e];
  size_t es = (size_t)EE * 64, ns = (size_t)NN * 64;
  #pragma unroll
  for (int i = 0; i < 8; ++i)
    atomicAdd(&agg[(size_t)i*ns + (size_t)r*64 + c], msg[(size_t)i*es + idx]);
}
__global__ void k_div_agg(float* __restrict__ agg, const float* __restrict__ cnt) {
  int idx = blockIdx.x * 256 + threadIdx.x;
  if (idx >= NN * 64) return;
  float d = fmaxf(cnt[idx >> 6], 1.0f);
  size_t ns = (size_t)NN * 64;
  #pragma unroll
  for (int i = 0; i < 8; ++i) agg[(size_t)i * ns + idx] /= d;
}
__global__ void k_build_node(const float* __restrict__ h, const float* __restrict__ agg,
                             float* __restrict__ Xn) {
  int idx = blockIdx.x * 256 + threadIdx.x;
  if (idx >= NN * KP_NODE) return;
  int n = idx / KP_NODE, c = idx % KP_NODE;
  size_t ns = (size_t)NN * 64, xs = (size_t)NN * KP_NODE;
  #pragma unroll
  for (int i = 0; i < 8; ++i) {
    float v = (c < 64) ? h[(size_t)i*ns + (size_t)n*64 + c]
                       : agg[(size_t)i*ns + (size_t)n*64 + (c - 64)];
    Xn[(size_t)i * xs + idx] = v;
  }
}
__global__ void k_resid(float* __restrict__ h, const float* __restrict__ o, int n) {
  int i = blockIdx.x * 256 + threadIdx.x;
  if (i < n) h[i] += o[i];
}

// ---------------- readout ----------------
__global__ void k_pool(const float* __restrict__ s, const int* __restrict__ batch,
                       float* __restrict__ pooled) {
  int idx = blockIdx.x * 256 + threadIdx.x;
  if (idx >= NN * 64) return;
  atomicAdd(&pooled[(size_t)batch[idx >> 6] * 64 + (idx & 63)], s[idx]);
}
__global__ void k_head(const float* __restrict__ pooled, const float* __restrict__ w1,
                       const float* __restrict__ b1, const float* __restrict__ w2,
                       const float* __restrict__ b2, float* __restrict__ out) {
  __shared__ float sp[64], red[64];
  int g = blockIdx.x, c = threadIdx.x;
  sp[c] = pooled[(size_t)g * 64 + c];
  __syncthreads();
  float acc = b1[c];
  #pragma unroll 8
  for (int m = 0; m < 64; ++m) acc += sp[m] * w1[c * 64 + m];
  float hid = acc * sigm(acc);          // silu
  red[c] = hid * w2[c];
  __syncthreads();
  for (int s = 32; s > 0; s >>= 1) { if (c < s) red[c] += red[c + s]; __syncthreads(); }
  if (c == 0) out[g] = red[0] + b2[0];
}

// ======================= host orchestration =======================
extern "C" void kernel_launch(void* const* d_in, const int* in_sizes, int n_in,
                              void* d_out, int out_size, void* d_ws, size_t ws_size,
                              hipStream_t stream) {
  (void)in_sizes; (void)out_size; (void)ws_size;
  // -------- input map (jax pytree: dicts flattened by sorted keys) --------
  const float* P[200];
  for (int i = 0; i < n_in && i < 200; ++i) P[i] = (const float*)d_in[i];
  const float* pos      = P[0];
  const float* emb      = P[1];                  // atom_embed [100][64]
  const float* inproj_b = P[2];                  // [64]
  const float* inproj_w = P[3];                  // [64][65]
  // layers occupy 4 .. 163 (4 layers x 4 blocks x 10 leaves)
  const float* post1_b  = P[164];
  const float* post1_w  = P[165];
  const float* post2_b  = P[166];
  const float* post2_w  = P[167];
  const float* prep_b   = P[168];
  const float* prep_w   = P[169];
  const int* z     = (const int*)d_in[170];
  const int* batch = (const int*)d_in[171];
  const int* ei    = (const int*)d_in[172];
  const int* rows = ei, *cols = ei + EE;

  // -------- workspace carve-up --------
  char* cur = (char*)d_ws;
  auto alloc = [&](size_t nfl) -> float* {
    float* p = (float*)cur;
    cur += ((nfl * sizeof(float) + 255) / 256) * 256;
    return p;
  };
  float* bufA  = alloc((size_t)8 * EE * KP_EDGE);  // edge input / block outputs / node input
  float* bufB  = alloc((size_t)8 * EE * 64);       // t (after lin+silu)
  float* bufC  = alloc((size_t)8 * EE * 64);       // xr
  float* h     = alloc((size_t)8 * NN * 64);
  float* x0    = alloc((size_t)8 * NN * KP_X0);
  float* agg   = alloc((size_t)8 * NN * 64);
  float* rbf   = alloc((size_t)EE * NRBF);
  float* cntN  = alloc(NN);
  float* cntG  = alloc(GG);
  float* psum  = alloc(GG * 3);
  float* pmean = alloc(GG * 3);
  float* ppool = alloc((size_t)NN * 64);
  float* pooled= alloc((size_t)GG * 64);
  unsigned short* wbuf = (unsigned short*)alloc((size_t)4 * KP_NODE * 64 / 2 + 64);

  auto Z = [&](float* p, int n) { k_zero<<<cdiv(n, 256), 256, 0, stream>>>(p, n); };

  // gemm wrapper: prep weights (bf16 [4][KP][64]) then WMMA GEMM
  auto gemm = [&](const float* X, const float* Wsrc, const float* bias, float* Out,
                  int M, int K, int KP, bool subspaces, int nblades) {
    int tw = 4 * KP * 64;
    k_prepw<<<cdiv(tw, 256), 256, 0, stream>>>(Wsrc, wbuf, K, KP, subspaces ? 1 : 0);
    dim3 grid(cdiv(M, 256), 1, nblades);
    k_gemm<<<grid, 256, 0, stream>>>(X, wbuf, bias, Out, M, KP);
  };

  // one cemlp block (in-place on `inout`, scratch t/xr)
  auto block = [&](float* inout, float* t, float* xr, int B, int K, int KP, int pb) {
    const float* b_left = P[pb + 0];
    const float* gp_w   = P[pb + 1];
    const float* lin_b  = P[pb + 2];
    const float* lin_w  = P[pb + 3];
    const float* ln_a   = P[pb + 4];
    const float* nrm_a  = P[pb + 5];
    const float* sa     = P[pb + 6];
    const float* sb     = P[pb + 7];
    const float* w_left = P[pb + 8];
    const float* w_rt   = P[pb + 9];
    int ew = cdiv(B * 64, 256);
    gemm(inout, lin_w, lin_b, t, B, K, KP, true, 8);        // mv_linear
    k_silu<<<ew, 256, 0, stream>>>(t, sa, sb, B);           // mv_silu
    gemm(t, w_rt, nullptr, xr, B, 64, 64, true, 8);         // w_right
    k_mvnorm<<<ew, 256, 0, stream>>>(xr, nrm_a, B);         // mv_normalize
    gemm(t, w_left, b_left, inout, B, 64, 64, true, 8);     // left (+bias)
    k_geo<<<ew, 256, 0, stream>>>(t, xr, inout, gp_w, B);   // geo, (left+geo)/sqrt2
    k_ln<<<B, 64, 0, stream>>>(inout, ln_a, B);             // mv_layernorm
  };

  // -------- preprocessing --------
  Z(cntN, NN); Z(cntG, GG); Z(psum, GG * 3);
  k_count_edges<<<cdiv(EE, 256), 256, 0, stream>>>(rows, cntN);
  k_nodestats<<<cdiv(NN, 256), 256, 0, stream>>>(pos, batch, cntG, psum);
  k_pmean<<<cdiv(GG * 3, 256), 256, 0, stream>>>(psum, cntG, pmean);
  k_rbf<<<cdiv(EE, 256), 256, 0, stream>>>(pos, rows, cols, rbf);
  k_x0<<<cdiv(NN * KP_X0, 256), 256, 0, stream>>>(emb, z, pos, pmean, batch, x0);
  gemm(x0, inproj_w, inproj_b, h, NN, 65, KP_X0, /*subspaces=*/false, 8);  // in_proj

  // -------- EGCL layers --------
  for (int L = 0; L < NLAYER; ++L) {
    int base = 4 + 40 * L;
    // edge model
    k_build_edge<<<cdiv(EE * KP_EDGE, 256), 256, 0, stream>>>(h, rbf, rows, cols, bufA);
    block(bufA, bufB, bufC, EE, 84, KP_EDGE, base + 0);
    block(bufA, bufB, bufC, EE, 64, 64,      base + 10);
    // aggregate (segment mean over rows)
    Z(agg, 8 * NN * 64);
    k_scatter_msg<<<cdiv(EE * 64, 256), 256, 0, stream>>>(bufA, rows, agg);
    k_div_agg<<<cdiv(NN * 64, 256), 256, 0, stream>>>(agg, cntN);
    // node model
    k_build_node<<<cdiv(NN * KP_NODE, 256), 256, 0, stream>>>(h, agg, bufA);
    block(bufA, bufB, bufC, NN, 128, KP_NODE, base + 20);
    block(bufA, bufB, bufC, NN, 64, 64,       base + 30);
    k_resid<<<cdiv(8 * NN * 64, 256), 256, 0, stream>>>(h, bufA, 8 * NN * 64);
  }

  // -------- readout --------
  gemm(h, prep_w, prep_b, ppool, NN, 64, 64, true, /*nblades=*/1);  // scalar blade only
  Z(pooled, GG * 64);
  k_pool<<<cdiv(NN * 64, 256), 256, 0, stream>>>(ppool, batch, pooled);
  k_head<<<GG, 64, 0, stream>>>(pooled, post1_w, post1_b, post2_w, post2_b,
                                (float*)d_out);
}
